// TriangleAttention_55851754717711
// MI455X (gfx1250) — compile-verified, matrix operand order checked by reading, not run on hardware
//
#include <hip/hip_runtime.h>
#include <hip/hip_bf16.h>

// ---------------------------------------------------------------------------
// Triangle attention for MI455X (gfx1250, wave32, WMMA 16x16x32 f16).
// L=256, D_PAIR=128, N_HEAD=4, D_HID=32, NPOS=65536.
// ---------------------------------------------------------------------------

#define LVAL   256
#define DPAIR  128
#define NHEAD  4
#define DHID   32
#define NPOS   (LVAL * LVAL)
#define SCALING 0.17677669529663687f   // 1/sqrt(32)

typedef __attribute__((ext_vector_type(16))) _Float16 v16h;
typedef __attribute__((ext_vector_type(8)))  _Float16 v8h;
typedef __attribute__((ext_vector_type(8)))  float    v8f;
typedef int vs4i __attribute__((vector_size(16)));   // async-LDS builtin arg type

#if __has_builtin(__builtin_amdgcn_global_load_async_to_lds_b128) && \
    __has_builtin(__builtin_amdgcn_s_wait_asynccnt)
#define HAVE_ASYNC_LDS 1
#else
#define HAVE_ASYNC_LDS 0
#endif

// Load one 16x32 f16 fragment (A-matrix layout; also used for B as B^T rows)
// from row-major storage with row stride `ld` (elements).
// ISA layout (16-bit A 16x32): lanes 0-15 -> M=0-15, cols {0..7,16..23};
// lanes 16-31 -> M=0-15, cols {8..15,24..31}. Two b128 loads per lane.
__device__ inline v16h load_frag16x32(const _Float16* p, int ld) {
  const int lane = threadIdx.x & 31;
  const int row  = lane & 15;
  const int hi   = lane >> 4;
  const _Float16* r = p + row * ld + hi * 8;
  v8h a = *(const v8h*)(r);
  v8h b = *(const v8h*)(r + 16);
  v16h f;
#pragma unroll
  for (int t = 0; t < 8; ++t) { f[t] = a[t]; f[t + 8] = b[t]; }
  return f;
}

__device__ inline v8f wmma_f16(v16h a, v16h b, v8f c) {
  return __builtin_amdgcn_wmma_f32_16x16x32_f16(false, a, false, b,
                                                (short)0, c, false, false);
}

// ---------------------------------------------------------------------------
// Kernel 1: LayerNorm over D=128, f32 in -> f16 out. One wave per position.
// ---------------------------------------------------------------------------
__global__ __launch_bounds__(256) void ln_kernel(
    const float* __restrict__ pair, const float* __restrict__ gamma,
    const float* __restrict__ beta, _Float16* __restrict__ xn) {
  const int wave = threadIdx.x >> 5;
  const int lane = threadIdx.x & 31;
  const size_t pos = (size_t)blockIdx.x * 8 + wave;
  const float4 xv = *(const float4*)(pair + pos * DPAIR + lane * 4);
  float s  = xv.x + xv.y + xv.z + xv.w;
  float ss = xv.x * xv.x + xv.y * xv.y + xv.z * xv.z + xv.w * xv.w;
#pragma unroll
  for (int m = 1; m < 32; m <<= 1) {
    s  += __shfl_xor(s,  m, 32);
    ss += __shfl_xor(ss, m, 32);
  }
  const float mu  = s * (1.0f / DPAIR);
  const float var = ss * (1.0f / DPAIR) - mu * mu;
  const float inv = rsqrtf(var + 1e-5f);
  const float4 gv = *(const float4*)(gamma + lane * 4);
  const float4 bv = *(const float4*)(beta  + lane * 4);
  _Float16* o = xn + pos * DPAIR + lane * 4;
  o[0] = (_Float16)((xv.x - mu) * inv * gv.x + bv.x);
  o[1] = (_Float16)((xv.y - mu) * inv * gv.y + bv.y);
  o[2] = (_Float16)((xv.z - mu) * inv * gv.z + bv.z);
  o[3] = (_Float16)((xv.w - mu) * inv * gv.w + bv.w);
}

// ---------------------------------------------------------------------------
// Kernel 2: projection GEMM (65536x128)x(128x128), specialized per output.
// MODE 0: q (scale by 1/sqrt(32)); MODE 1: plain (k, v); MODE 2: sigmoid gate.
// grid.x = NPOS/128; block = 256 (8 waves, 16 rows each). W^T f16 in LDS.
// ---------------------------------------------------------------------------
#define WT_LD 136   // 128 + 8 pad (272B row stride, 16B aligned)
template <int MODE>
__global__ __launch_bounds__(256) void proj_kernel(
    const _Float16* __restrict__ xn, const float* __restrict__ W,
    const float* __restrict__ bg, _Float16* __restrict__ dst) {
  __shared__ _Float16 WT[128 * WT_LD];
  for (int idx = threadIdx.x; idx < 128 * 128; idx += 256) {
    const int kk = idx >> 7, n = idx & 127;
    WT[n * WT_LD + kk] = (_Float16)W[idx];     // transpose to N-major
  }
  __syncthreads();

  const int wave = threadIdx.x >> 5;
  const int lane = threadIdx.x & 31;
  const int kcol = lane & 15, hi = lane >> 4;
  const size_t m0 = (size_t)blockIdx.x * 128 + wave * 16;

  v16h A[4];
#pragma unroll
  for (int kk = 0; kk < 4; ++kk)
    A[kk] = load_frag16x32(xn + m0 * DPAIR + kk * 32, DPAIR);

#pragma unroll
  for (int nt = 0; nt < 8; ++nt) {
    v8f acc = {};
#pragma unroll
    for (int kk = 0; kk < 4; ++kk) {
      v16h B = load_frag16x32(WT + (nt * 16) * WT_LD + kk * 32, WT_LD);
      acc = wmma_f16(A[kk], B, acc);
    }
    const int n = nt * 16 + kcol;
    const float bgn = (MODE == 2) ? bg[n] : 0.0f;
#pragma unroll
    for (int r = 0; r < 8; ++r) {
      const size_t m = m0 + r + 8 * hi;
      float val = acc[r];
      if (MODE == 0) val *= SCALING;
      if (MODE == 2) val = 1.0f / (1.0f + __expf(-(val + bgn)));
      dst[m * DPAIR + n] = (_Float16)val;
    }
  }
}

// ---------------------------------------------------------------------------
// Kernel 3: bias = xn @ Wb (128 -> 4), stored as per-head planes
// biasws[h][j][k] so the attention kernel's reads are fully coalesced.
// Thread t: h = t >> 16, pos = t & 65535 (consecutive threads -> same h,
// consecutive pos -> coalesced reads and writes).
// ---------------------------------------------------------------------------
__global__ __launch_bounds__(256) void bias_kernel(
    const _Float16* __restrict__ xn, const float* __restrict__ Wb,
    float* __restrict__ biasws) {
  const size_t t = (size_t)blockIdx.x * blockDim.x + threadIdx.x;
  const int h = (int)(t >> 16);          // NPOS == 65536
  const size_t pos = t & (NPOS - 1);
  const _Float16* x = xn + pos * DPAIR;
  float acc = 0.0f;
#pragma unroll 8
  for (int c = 0; c < DPAIR; ++c) acc += (float)x[c] * Wb[c * NHEAD + h];
  biasws[(size_t)h * NPOS + pos] = acc;
}

// ---------------------------------------------------------------------------
// Kernel 4: attention with ONLINE (flash-style) softmax (no VGPR spills).
// grid = (2, 256): block = (j-slab of 128, row i). Per head: K_h staged via
// async direct-to-LDS DMA (GLOBAL_LOAD_ASYNC_TO_LDS_B128, ASYNCcnt);
// V_h^T staged manually (transpose). Each wave owns 16 query rows; keys in
// 32-wide chunks: 2 WMMAs for scores, bias + running max/sum rescale, exp
// tile -> wave-private 1KB LDS re-tile, 2 WMMAs for P.V.
// ---------------------------------------------------------------------------
#define VT_LD 264   // 256 + 8 pad
__global__ __launch_bounds__(256) void attn_kernel(
    const _Float16* __restrict__ q, const _Float16* __restrict__ k,
    const _Float16* __restrict__ v, const float* __restrict__ biasws,
    const _Float16* __restrict__ gate, _Float16* __restrict__ att) {
  __shared__ _Float16 Ksh[LVAL * DHID];       // 16 KB
  __shared__ _Float16 VTsh[DHID * VT_LD];     // ~16.5 KB
  __shared__ _Float16 Pbuf[8][16 * 32];       // 8 KB (wave-private tiles)

  const int i     = blockIdx.y;
  const int jhalf = blockIdx.x;
  const int wave  = threadIdx.x >> 5;
  const int lane  = threadIdx.x & 31;
  const int kcol  = lane & 15, hi = lane >> 4;
  const int j0    = jhalf * 128 + wave * 16;

  for (int h = 0; h < NHEAD; ++h) {
    __syncthreads();
    // --- K rows: async DMA straight into LDS (one 64B row per thread) ---
    {
      const _Float16* srcK =
          k + ((size_t)(i * LVAL + threadIdx.x)) * DPAIR + h * DHID;
      _Float16* dstK = Ksh + threadIdx.x * DHID;
#if HAVE_ASYNC_LDS
#pragma unroll
      for (int t = 0; t < 4; ++t)
        __builtin_amdgcn_global_load_async_to_lds_b128(
            (vs4i*)(srcK + t * 8), (vs4i*)(dstK + t * 8), 0, 0);
#else
#pragma unroll
      for (int t = 0; t < 4; ++t)
        *(v8h*)(dstK + t * 8) = *(const v8h*)(srcK + t * 8);
#endif
    }
    // --- V^T: manual transpose staging ---
    for (int idx = threadIdx.x; idx < LVAL * DHID; idx += 256) {
      const int kk = idx >> 5, c = idx & 31;
      VTsh[c * VT_LD + kk] =
          v[((size_t)(i * LVAL + kk)) * DPAIR + h * DHID + c];
    }
    // Prefetch next head's K/V rows (emits global_prefetch_b8).
    if (h + 1 < NHEAD) {
      const size_t nsrc =
          ((size_t)(i * LVAL + threadIdx.x)) * DPAIR + (h + 1) * DHID;
      __builtin_prefetch(k + nsrc, 0, 1);
      __builtin_prefetch(v + nsrc, 0, 1);
    }
#if HAVE_ASYNC_LDS
    __builtin_amdgcn_s_wait_asynccnt(0);
#endif
    __syncthreads();

    const v16h aQ =
        load_frag16x32(q + ((size_t)(i * LVAL + j0)) * DPAIR + h * DHID, DPAIR);
    const float* biasH = biasws + (size_t)h * NPOS;

    v8f O0 = {}, O1 = {};
    float m[8], sum[8];
#pragma unroll
    for (int r = 0; r < 8; ++r) { m[r] = -3.0e38f; sum[r] = 0.0f; }

    for (int c = 0; c < 8; ++c) {        // 32 keys per chunk
      // --- scores for this chunk: two 16x16 tiles ---
      const v16h bK0 = load_frag16x32(Ksh + (2 * c + 0) * 16 * DHID, DHID);
      const v16h bK1 = load_frag16x32(Ksh + (2 * c + 1) * 16 * DHID, DHID);
      v8f z = {};
      v8f S0 = wmma_f16(aQ, bK0, z);
      v8f S1 = wmma_f16(aQ, bK1, z);

      // --- bias + online softmax update ---
#pragma unroll
      for (int r = 0; r < 8; ++r) {
        const int j = j0 + r + 8 * hi;
        const float* bp = biasH + (size_t)j * LVAL + c * 32 + kcol;
        const float v0 = S0[r] + bp[0];    // coalesced: 16 lanes x 4B
        const float v1 = S1[r] + bp[16];
        float tm = fmaxf(v0, v1);
#pragma unroll
        for (int msk = 1; msk < 16; msk <<= 1)
          tm = fmaxf(tm, __shfl_xor(tm, msk, 32));
        const float mnew = fmaxf(m[r], tm);
        const float cf   = __expf(m[r] - mnew);
        m[r] = mnew;
        const float e0 = __expf(v0 - mnew);
        const float e1 = __expf(v1 - mnew);
        float ps = e0 + e1;
#pragma unroll
        for (int msk = 1; msk < 16; msk <<= 1) ps += __shfl_xor(ps, msk, 32);
        sum[r] = sum[r] * cf + ps;
        O0[r] *= cf;
        O1[r] *= cf;
        const int row = r + 8 * hi;
        Pbuf[wave][row * 32 + kcol]      = (_Float16)e0;
        Pbuf[wave][row * 32 + 16 + kcol] = (_Float16)e1;
      }
      // Pbuf is wave-private; same-wave DS ordering suffices (no barrier).

      // --- accumulate O += P(16x32) @ V(32x32) ---
      const v16h aP  = load_frag16x32(&Pbuf[wave][0], 32);
      const v16h bV0 = load_frag16x32(VTsh + 0  * VT_LD + c * 32, VT_LD);
      const v16h bV1 = load_frag16x32(VTsh + 16 * VT_LD + c * 32, VT_LD);
      O0 = wmma_f16(aP, bV0, O0);
      O1 = wmma_f16(aP, bV1, O1);
    }

    // --- normalize, gate, store ---
#pragma unroll
    for (int r = 0; r < 8; ++r) {
      const int j = j0 + r + 8 * hi;
      const size_t pos = (size_t)(i * LVAL + j);
      const float inv = 1.0f / sum[r];
      {
        const int n = kcol;
        const float g = (float)gate[pos * DPAIR + h * DHID + n];
        att[pos * DPAIR + h * DHID + n] = (_Float16)(O0[r] * inv * g);
      }
      {
        const int n = 16 + kcol;
        const float g = (float)gate[pos * DPAIR + h * DHID + n];
        att[pos * DPAIR + h * DHID + n] = (_Float16)(O1[r] * inv * g);
      }
    }
  }
}

// ---------------------------------------------------------------------------
// Kernel 5: out = att @ Wo + bo, f32 out. Same shape as proj_kernel.
// ---------------------------------------------------------------------------
__global__ __launch_bounds__(256) void out_kernel(
    const _Float16* __restrict__ att, const float* __restrict__ Wo,
    const float* __restrict__ bo, float* __restrict__ out) {
  __shared__ _Float16 WT[128 * WT_LD];
  for (int idx = threadIdx.x; idx < 128 * 128; idx += 256) {
    const int kk = idx >> 7, n = idx & 127;
    WT[n * WT_LD + kk] = (_Float16)Wo[idx];
  }
  __syncthreads();

  const int wave = threadIdx.x >> 5;
  const int lane = threadIdx.x & 31;
  const int kcol = lane & 15, hi = lane >> 4;
  const size_t m0 = (size_t)blockIdx.x * 128 + wave * 16;

  v16h A[4];
#pragma unroll
  for (int kk = 0; kk < 4; ++kk)
    A[kk] = load_frag16x32(att + m0 * DPAIR + kk * 32, DPAIR);

#pragma unroll
  for (int nt = 0; nt < 8; ++nt) {
    v8f acc = {};
#pragma unroll
    for (int kk = 0; kk < 4; ++kk) {
      v16h B = load_frag16x32(WT + (nt * 16) * WT_LD + kk * 32, WT_LD);
      acc = wmma_f16(A[kk], B, acc);
    }
    const int n = nt * 16 + kcol;
#pragma unroll
    for (int r = 0; r < 8; ++r) {
      const size_t m = m0 + r + 8 * hi;
      out[m * DPAIR + n] = acc[r] + bo[n];
    }
  }
}

// ---------------------------------------------------------------------------
// Launch.  Inputs: 0 pair, 1 gamma, 2 beta, 3 Wq, 4 Wk, 5 Wv, 6 Wb, 7 Wg,
// 8 bg, 9 Wo, 10 bo.  Output: f32 (1,256,256,128).
// Workspace (~102 MB): xn | q | k | v | gate | att (f16, 16 MB each) | bias.
// ---------------------------------------------------------------------------
extern "C" void kernel_launch(void* const* d_in, const int* in_sizes, int n_in,
                              void* d_out, int out_size, void* d_ws,
                              size_t ws_size, hipStream_t stream) {
  const float* pair  = (const float*)d_in[0];
  const float* gamma = (const float*)d_in[1];
  const float* beta  = (const float*)d_in[2];
  const float* Wq    = (const float*)d_in[3];
  const float* Wk    = (const float*)d_in[4];
  const float* Wv    = (const float*)d_in[5];
  const float* Wb    = (const float*)d_in[6];
  const float* Wg    = (const float*)d_in[7];
  const float* bg    = (const float*)d_in[8];
  const float* Wo    = (const float*)d_in[9];
  const float* bo    = (const float*)d_in[10];
  float* out = (float*)d_out;

  char* ws = (char*)d_ws;
  const size_t SEG = (size_t)NPOS * DPAIR * sizeof(_Float16);  // 16 MB
  _Float16* xn   = (_Float16*)(ws + 0 * SEG);
  _Float16* qbuf = (_Float16*)(ws + 1 * SEG);
  _Float16* kbuf = (_Float16*)(ws + 2 * SEG);
  _Float16* vbuf = (_Float16*)(ws + 3 * SEG);
  _Float16* gbuf = (_Float16*)(ws + 4 * SEG);
  _Float16* abuf = (_Float16*)(ws + 5 * SEG);
  float*    bbuf = (float*)   (ws + 6 * SEG);  // NHEAD*NPOS f32 = 1 MB

  ln_kernel<<<NPOS / 8, 256, 0, stream>>>(pair, gamma, beta, xn);
  proj_kernel<0><<<NPOS / 128, 256, 0, stream>>>(xn, Wq, nullptr, qbuf);
  proj_kernel<1><<<NPOS / 128, 256, 0, stream>>>(xn, Wk, nullptr, kbuf);
  proj_kernel<1><<<NPOS / 128, 256, 0, stream>>>(xn, Wv, nullptr, vbuf);
  proj_kernel<2><<<NPOS / 128, 256, 0, stream>>>(xn, Wg, bg, gbuf);
  bias_kernel<<<(NPOS * NHEAD) / 256, 256, 0, stream>>>(xn, Wb, bbuf);
  attn_kernel<<<dim3(2, LVAL), 256, 0, stream>>>(qbuf, kbuf, vbuf, bbuf, gbuf,
                                                 abuf);
  out_kernel<<<NPOS / 128, 256, 0, stream>>>(abuf, Wo, bo, out);
}
